// MultiLayerLSTM_58119497450297
// MI455X (gfx1250) — matrix operand order repaired and checked by reading, not compiled, not run on hardware
//
#include <hip/hip_runtime.h>

// ---------------------------------------------------------------------------
// 2-layer LSTM, persistent-kernel design for MI455X (gfx1250, wave32, WMMA).
// B=256,T=1024,E=64,H=128,V=27. 16 workgroups x 256 threads (8 waves).
// Each workgroup owns a 16-row batch tile and iterates all 1024 timesteps.
// Matmuls via v_wmma_f32_16x16x32_bf16. wx0/wh0/why LDS-resident (bf16,
// fragment-major), staged via GLOBAL_LOAD_ASYNC_TO_LDS_B128 (ASYNCcnt);
// wx1/wh1 streamed from L2 each step.
// ---------------------------------------------------------------------------

typedef __bf16  v16bf __attribute__((ext_vector_type(16)));
typedef float   v8f   __attribute__((ext_vector_type(8)));

#define TSTEPS 1024
#define VOCAB  27
#define EDIM   64
#define HDIM   128
#define GDIM   512   // 4*H

// ---- LDS layout (bytes) ----------------------------------------------------
#define OFF_WX0   0        //  64 KB : 32 ntiles * 2 kfrags * 1024B
#define OFF_WH0   65536    // 128 KB : 32 * 4 * 1024B
#define OFF_WHY   196608   //   8 KB : 2 * 4 * 1024B
#define OFF_EMB   204800   // 27*64 bf16 = 3456 B (row stride 128B)
#define OFF_B0    208384   // 512 f32
#define OFF_B1    210432   // 512 f32
#define OFF_BY    212480   // 27 f32 (padded)
#define OFF_GATES 212608   // 16*512 f32 = 32 KB
#define OFF_H0    245376   // 16*128 bf16 = 4 KB
#define OFF_H1    249472   // 4 KB
#define OFF_C0    253568   // 16*128 f32 = 8 KB
#define OFF_C1    261760   // 8 KB
#define SMEM_BYTES 269952  // ~264 KB < 320 KB/WGP

// ---- workspace layout (packed bf16 weights, fragment-major) ----------------
#define WS_WX0 0          //  65536 B  (NK=2, NT=32)
#define WS_WH0 65536      // 131072 B  (NK=4, NT=32)
#define WS_WX1 196608     // 131072 B
#define WS_WH1 327680     // 131072 B
#define WS_WHY 458752     //   8192 B  (NK=4, NT=2)

__device__ __forceinline__ unsigned short f2bf(float f) {
  unsigned int u = __float_as_uint(f);
  unsigned int r = u + 0x7FFFu + ((u >> 16) & 1u);   // round-to-nearest-even
  return (unsigned short)(r >> 16);
}

__device__ __forceinline__ float sigf(float x) {
  return 1.0f / (1.0f + __expf(-x));
}
__device__ __forceinline__ float tanhfast(float x) {
  return 2.0f * sigf(2.0f * x) - 1.0f;               // tanh via exp
}

// ---- CDNA5 async global->LDS copy (16B per lane, tracked by ASYNCcnt) ------
// §15.18.3 GLOBAL_LOAD_ASYNC_TO_LDS_B128: VDST = LDS byte address VGPR,
// VADDR = 64-bit global address VGPR pair, no SADDR ("off").
__device__ __forceinline__ void async_cp_b128(unsigned lds_addr,
                                              const char* gsrc) {
  unsigned long long ga = (unsigned long long)gsrc;
  asm volatile("global_load_async_to_lds_b128 %0, %1, off"
               :: "v"(lds_addr), "v"(ga) : "memory");
}
__device__ __forceinline__ void wait_async0() {
#if __has_builtin(__builtin_amdgcn_s_wait_asynccnt)
  __builtin_amdgcn_s_wait_asynccnt(0);
#else
  asm volatile("s_wait_asynccnt 0x0" ::: "memory");
#endif
}

// B fragment (32x16 bf16): lane's 16 bf16 are contiguous 32 bytes.
__device__ __forceinline__ v16bf lds_bfrag(const char* smem, int off, int lane) {
  union { uint4 q[2]; v16bf v; } u;
  const uint4* p = (const uint4*)(smem + off + lane * 32);
  u.q[0] = p[0];
  u.q[1] = p[1];
  return u.v;
}
__device__ __forceinline__ v16bf glb_bfrag(const char* base, int frag, int lane) {
  union { uint4 q[2]; v16bf v; } u;
  const uint4* p = (const uint4*)(base + frag * 1024 + lane * 32);
  u.q[0] = p[0];
  u.q[1] = p[1];
  return u.v;
}

// A fragment (16x32 bf16) from row-major bf16 tile in LDS.
// ISA 7.12.2: lanes 0-15 hold K octets [0..8) and [16..24); lanes 16-31 hold
// [8..16) and [24..32).  Two 16B ds reads per lane.
__device__ __forceinline__ v16bf lds_afrag(const char* smem, int off,
                                           int row_stride, int row, int kbase,
                                           int lane) {
  int koct = (lane & 16) ? 8 : 0;
  const char* p = smem + off + row * row_stride + (kbase + koct) * 2;
  union { uint4 q[2]; v16bf v; } u;
  u.q[0] = *(const uint4*)p;
  u.q[1] = *(const uint4*)(p + 32);   // +16 K elements
  return u.v;
}

__device__ __forceinline__ v8f wmma_bf16(v16bf a, v16bf b, v8f c) {
  return __builtin_amdgcn_wmma_f32_16x16x32_bf16(false, a, false, b,
                                                 (short)0, c, false, false);
}

// ---------------------------------------------------------------------------
// Prep: pack f32 weight [K][Nsrc] into fragment-major bf16.
// frag id = ntile*NK + ktile; within a frag: lane-major blocks of 32 bytes,
// dword v holds {W[k][n], W[k+1][n]} with k = ik*32 + 2(v&3) + 16*(v>=4) + 8*(lane>=16),
// n = in*16 + (lane&15).  This is exactly the per-lane B-operand order.
// ---------------------------------------------------------------------------
__global__ void pack_wmma_b(const float* __restrict__ src,
                            unsigned int* __restrict__ dst,
                            int Nsrc, int NK, int NT) {
  int d = blockIdx.x * blockDim.x + threadIdx.x;
  int total = NT * NK * 256;
  if (d >= total) return;
  int frag = d >> 8;
  int rem  = d & 255;
  int lane = rem >> 3;
  int v    = rem & 7;
  int in   = frag / NK;
  int ik   = frag - in * NK;
  int k = ik * 32 + 2 * (v & 3) + ((v >> 2) << 4) + ((lane >> 4) << 3);
  int n = in * 16 + (lane & 15);
  float a = (n < Nsrc) ? src[k * Nsrc + n] : 0.0f;
  float b = (n < Nsrc) ? src[(k + 1) * Nsrc + n] : 0.0f;
  dst[d] = (unsigned int)f2bf(a) | ((unsigned int)f2bf(b) << 16);
}

// ---------------------------------------------------------------------------
// Persistent LSTM kernel: 16 blocks (batch tiles of 16) x 256 threads (8 waves)
// ---------------------------------------------------------------------------
__global__ void __launch_bounds__(256)
lstm_persist(const int* __restrict__ x, const float* __restrict__ emb,
             const float* __restrict__ b0g, const float* __restrict__ b1g,
             const float* __restrict__ byg, const char* __restrict__ wpk,
             float* __restrict__ out) {
  extern __shared__ char smem[];
  const int tid   = threadIdx.x;
  const int lane  = tid & 31;
  const int wave  = tid >> 5;
  const int bbase = blockIdx.x * 16;

  // ---- stage LDS-resident packed weights via async global->LDS DMA ---------
  // wx0+wh0 are contiguous (192 KB) in both workspace and LDS; why is 8 KB.
  // 48+2 async issues per wave, well under the 63-entry ASYNCcnt.
  {
    unsigned lds_base = (unsigned)(unsigned long long)smem;  // LDS aperture low bits
    for (int i = tid; i < (65536 + 131072) / 16; i += 256)
      async_cp_b128(lds_base + OFF_WX0 + i * 16, wpk + WS_WX0 + i * 16);
    for (int i = tid; i < 8192 / 16; i += 256)
      async_cp_b128(lds_base + OFF_WHY + i * 16, wpk + WS_WHY + i * 16);
  }
  // ---- overlap: embedding -> bf16 LDS, biases -> f32 LDS, zero state -------
  {
    unsigned short* el = (unsigned short*)(smem + OFF_EMB);
    for (int i = tid; i < VOCAB * EDIM; i += 256) el[i] = f2bf(emb[i]);
    float* b0l = (float*)(smem + OFF_B0);
    float* b1l = (float*)(smem + OFF_B1);
    for (int i = tid; i < GDIM; i += 256) { b0l[i] = b0g[i]; b1l[i] = b1g[i]; }
    float* byl = (float*)(smem + OFF_BY);
    for (int i = tid; i < VOCAB; i += 256) byl[i] = byg[i];
    unsigned short* h0l = (unsigned short*)(smem + OFF_H0);
    unsigned short* h1l = (unsigned short*)(smem + OFF_H1);
    float* c0l = (float*)(smem + OFF_C0);
    float* c1l = (float*)(smem + OFF_C1);
    for (int i = tid; i < 16 * HDIM; i += 256) {
      h0l[i] = 0; h1l[i] = 0; c0l[i] = 0.0f; c1l[i] = 0.0f;
    }
  }
  wait_async0();          // drain ASYNCcnt: all DMA'd weights resident in LDS
  __syncthreads();

  const char* gwx1 = wpk + WS_WX1;
  const char* gwh1 = wpk + WS_WH1;
  float* gts = (float*)(smem + OFF_GATES);
  float* b0l = (float*)(smem + OFF_B0);
  float* b1l = (float*)(smem + OFF_B1);
  float* byl = (float*)(smem + OFF_BY);
  float* c0l = (float*)(smem + OFF_C0);
  float* c1l = (float*)(smem + OFF_C1);
  unsigned short* h0l = (unsigned short*)(smem + OFF_H0);
  unsigned short* h1l = (unsigned short*)(smem + OFF_H1);

  const v8f zero8 = {0.f, 0.f, 0.f, 0.f, 0.f, 0.f, 0.f, 0.f};
  const int rowA = lane & 15;           // batch row within tile (A-matrix M)
  const int rowHi = (lane >> 4) << 3;   // C/D: lanes 16-31 hold M = r+8

  for (int t = 0; t < TSTEPS; ++t) {
    // ======================= layer 0: gates = xt@wx0 + h0@wh0 ==============
    int tok = x[(bbase + rowA) * TSTEPS + t];
    v16bf ax[2], ah[4];
#pragma unroll
    for (int kf = 0; kf < 2; ++kf)
      ax[kf] = lds_afrag(smem, OFF_EMB, EDIM * 2, tok, kf * 32, lane);
#pragma unroll
    for (int kf = 0; kf < 4; ++kf)
      ah[kf] = lds_afrag(smem, OFF_H0, HDIM * 2, rowA, kf * 32, lane);

    v8f acc[4];
#pragma unroll
    for (int j = 0; j < 4; ++j) acc[j] = zero8;
#pragma unroll
    for (int kf = 0; kf < 2; ++kf) {
#pragma unroll
      for (int j = 0; j < 4; ++j) {
        int nt = wave * 4 + j;
        v16bf b = lds_bfrag(smem, OFF_WX0 + (nt * 2 + kf) * 1024, lane);
        acc[j] = wmma_bf16(ax[kf], b, acc[j]);
      }
    }
#pragma unroll
    for (int kf = 0; kf < 4; ++kf) {
#pragma unroll
      for (int j = 0; j < 4; ++j) {
        int nt = wave * 4 + j;
        v16bf b = lds_bfrag(smem, OFF_WH0 + (nt * 4 + kf) * 1024, lane);
        acc[j] = wmma_bf16(ah[kf], b, acc[j]);
      }
    }
#pragma unroll
    for (int j = 0; j < 4; ++j) {
      int nt = wave * 4 + j;
#pragma unroll
      for (int r = 0; r < 8; ++r)
        gts[(r + rowHi) * GDIM + nt * 16 + (lane & 15)] = acc[j][r];
    }
    __syncthreads();

    // layer-0 elementwise: c0,h0 update (gate order f,i,g,o)
#pragma unroll
    for (int e = 0; e < 8; ++e) {
      int lin = e * 256 + tid;            // = row*128 + col
      int row = lin >> 7, col = lin & 127;
      float f = gts[row * GDIM + col]        + b0l[col];
      float i = gts[row * GDIM + 128 + col]  + b0l[128 + col];
      float g = gts[row * GDIM + 256 + col]  + b0l[256 + col];
      float o = gts[row * GDIM + 384 + col]  + b0l[384 + col];
      float c = sigf(f) * c0l[lin] + sigf(i) * tanhfast(g);
      c0l[lin] = c;
      h0l[lin] = f2bf(tanhfast(c) * sigf(o));
    }
    __syncthreads();

    // ======================= layer 1: gates = h0@wx1 + h1@wh1 ==============
    v16bf a0n[4], a1o[4];
#pragma unroll
    for (int kf = 0; kf < 4; ++kf) {
      a0n[kf] = lds_afrag(smem, OFF_H0, HDIM * 2, rowA, kf * 32, lane);
      a1o[kf] = lds_afrag(smem, OFF_H1, HDIM * 2, rowA, kf * 32, lane);
    }
#pragma unroll
    for (int j = 0; j < 4; ++j) acc[j] = zero8;
#pragma unroll
    for (int kf = 0; kf < 4; ++kf) {
#pragma unroll
      for (int j = 0; j < 4; ++j) {
        int nt = wave * 4 + j;
        v16bf b = glb_bfrag(gwx1, nt * 4 + kf, lane);   // streamed from L2
        acc[j] = wmma_bf16(a0n[kf], b, acc[j]);
      }
    }
#pragma unroll
    for (int kf = 0; kf < 4; ++kf) {
#pragma unroll
      for (int j = 0; j < 4; ++j) {
        int nt = wave * 4 + j;
        v16bf b = glb_bfrag(gwh1, nt * 4 + kf, lane);   // streamed from L2
        acc[j] = wmma_bf16(a1o[kf], b, acc[j]);
      }
    }
#pragma unroll
    for (int j = 0; j < 4; ++j) {
      int nt = wave * 4 + j;
#pragma unroll
      for (int r = 0; r < 8; ++r)
        gts[(r + rowHi) * GDIM + nt * 16 + (lane & 15)] = acc[j][r];
    }
    __syncthreads();

    // layer-1 elementwise: c1,h1 update
#pragma unroll
    for (int e = 0; e < 8; ++e) {
      int lin = e * 256 + tid;
      int row = lin >> 7, col = lin & 127;
      float f = gts[row * GDIM + col]        + b1l[col];
      float i = gts[row * GDIM + 128 + col]  + b1l[128 + col];
      float g = gts[row * GDIM + 256 + col]  + b1l[256 + col];
      float o = gts[row * GDIM + 384 + col]  + b1l[384 + col];
      float c = sigf(f) * c1l[lin] + sigf(i) * tanhfast(g);
      c1l[lin] = c;
      h1l[lin] = f2bf(tanhfast(c) * sigf(o));
    }
    __syncthreads();

    // ======================= output: y = h1@why + by (waves 0,1) ===========
    if (wave < 2) {                        // wave-uniform branch: EXEC all-1s
      v16bf a1[4];
#pragma unroll
      for (int kf = 0; kf < 4; ++kf)
        a1[kf] = lds_afrag(smem, OFF_H1, HDIM * 2, rowA, kf * 32, lane);
      v8f oacc = zero8;
#pragma unroll
      for (int kf = 0; kf < 4; ++kf) {
        v16bf b = lds_bfrag(smem, OFF_WHY + (wave * 4 + kf) * 1024, lane);
        oacc = wmma_bf16(a1[kf], b, oacc);
      }
      int col = wave * 16 + (lane & 15);
      if (col < VOCAB) {
        float bias = byl[col];
#pragma unroll
        for (int r = 0; r < 8; ++r) {
          int row = r + rowHi;
          out[((size_t)(bbase + row) * TSTEPS + t) * VOCAB + col] =
              oacc[r] + bias;
        }
      }
    }
  }
}

// ---------------------------------------------------------------------------
extern "C" void kernel_launch(void* const* d_in, const int* in_sizes, int n_in,
                              void* d_out, int out_size, void* d_ws,
                              size_t ws_size, hipStream_t stream) {
  (void)in_sizes; (void)n_in; (void)out_size; (void)ws_size;
  const int*   x   = (const int*)d_in[0];
  const float* emb = (const float*)d_in[1];
  const float* wx0 = (const float*)d_in[2];
  const float* wh0 = (const float*)d_in[3];
  const float* b0  = (const float*)d_in[4];
  const float* wx1 = (const float*)d_in[5];
  const float* wh1 = (const float*)d_in[6];
  const float* b1  = (const float*)d_in[7];
  const float* why = (const float*)d_in[8];
  const float* by  = (const float*)d_in[9];
  char*  wpk = (char*)d_ws;
  float* out = (float*)d_out;

  // pack weights -> fragment-major bf16 in workspace
  pack_wmma_b<<<64, 256, 0, stream>>>(wx0, (unsigned int*)(wpk + WS_WX0),
                                      512, 2, 32);
  pack_wmma_b<<<128, 256, 0, stream>>>(wh0, (unsigned int*)(wpk + WS_WH0),
                                       512, 4, 32);
  pack_wmma_b<<<128, 256, 0, stream>>>(wx1, (unsigned int*)(wpk + WS_WX1),
                                       512, 4, 32);
  pack_wmma_b<<<128, 256, 0, stream>>>(wh1, (unsigned int*)(wpk + WS_WH1),
                                       512, 4, 32);
  pack_wmma_b<<<8, 256, 0, stream>>>(why, (unsigned int*)(wpk + WS_WHY),
                                     VOCAB, 4, 2);

  // persistent LSTM: 16 batch tiles x 256 threads, 264 KB dynamic LDS
  lstm_persist<<<16, 256, SMEM_BYTES, stream>>>(x, emb, b0, b1, by, wpk, out);
}
